// BatchedGraphSAGEMean1_9337258902039
// MI455X (gfx1250) — matrix operand order repaired and checked by reading, not compiled, not run on hardware
//
#include <hip/hip_runtime.h>

// Problem constants (from reference): B=8, S=512, P=8, Fin=512, Out=512
#define BB   8
#define SS   512
#define PP   8
#define FF   512
#define DD   4096      // PP*FF
#define OUTC 512
#define C3   1536      // 3*Out
#define ROWS 32768     // BB*SS*PP
#define NBS  4096      // BB*SS

typedef __attribute__((ext_vector_type(16))) _Float16 v16h;
typedef __attribute__((ext_vector_type(8)))  float    v8f;

struct alignas(16) Q16 { unsigned int a, b, c, d; };
union FragU { v16h v; _Float16 h[16]; Q16 q[2]; };

// ---------------------------------------------------------------------------
// WMMA fragment loaders (wave32, 16x16x32 f16)
// A (16x32, MxK): lane L<16 -> row L, elems 0..7 = K[0..7], 8..15 = K[16..23];
//                 lane L>=16 -> row L-16, K offset +8 (ISA 7.12.2 table).
// B (32x16, KxN): lane L holds column N=L&15; elems 0..7 = K[hi*8 .. hi*8+7],
//                 8..15 = K[16+hi*8 .. 16+hi*8+7] (same K order as A).
// ---------------------------------------------------------------------------
__device__ __forceinline__ v16h load_fragA(const _Float16* __restrict__ tile, int ld) {
  const int lane = threadIdx.x & 31;
  const int r  = lane & 15;
  const int hi = lane >> 4;
  const _Float16* p = tile + (size_t)r * ld + hi * 8;
  FragU f;
  f.q[0] = *reinterpret_cast<const Q16*>(p);        // K k0+hi*8 .. +7
  f.q[1] = *reinterpret_cast<const Q16*>(p + 16);   // K k0+16+hi*8 .. +7
  return f.v;
}

__device__ __forceinline__ v16h load_fragB(const _Float16* __restrict__ tile, int ld) {
  const int lane = threadIdx.x & 31;
  const int n  = lane & 15;
  const int hi = lane >> 4;
  const _Float16* p = tile + (size_t)n * ld + hi * 8;
  FragU f;
  f.q[0] = *reinterpret_cast<const Q16*>(p);
  f.q[1] = *reinterpret_cast<const Q16*>(p + 16);
  return f.v;
}

__device__ __forceinline__ v8f wmma16(v16h a, v16h b, v8f c) {
  return __builtin_amdgcn_wmma_f32_16x16x32_f16(false, a, false, b, (short)0, c,
                                                false, false);
}

// ---------------------------------------------------------------------------
// Pass 1: f32 -> f16 copies of x (row-major + transposed [b][d][s]) + row norms
// ---------------------------------------------------------------------------
__global__ __launch_bounds__(256)
void convx_kernel(const float* __restrict__ x, _Float16* __restrict__ xh,
                  _Float16* __restrict__ xT, float* __restrict__ nrm) {
  const int s = blockIdx.x, b = blockIdx.y, tid = threadIdx.x;
  const float* xr = x + ((size_t)b * SS + s) * DD;
  _Float16* hr = xh + ((size_t)b * SS + s) * DD;
  _Float16* tb = xT + (size_t)b * DD * SS + s;
  float ss = 0.f;
#pragma unroll
  for (int c = 0; c < DD / 256; ++c) {
    int d = c * 256 + tid;
    float v = xr[d];
    ss += v * v;
    _Float16 h = (_Float16)v;
    hr[d] = h;
    tb[(size_t)d * SS] = h;
  }
  __shared__ float sm[256];
  sm[tid] = ss; __syncthreads();
  for (int st = 128; st > 0; st >>= 1) {
    if (tid < st) sm[tid] += sm[tid + st];
    __syncthreads();
  }
  if (tid == 0) nrm[b * SS + s] = sqrtf(sm[0]);
}

__global__ void convw_kernel(const float* __restrict__ w, _Float16* __restrict__ wh, int n) {
  int i = blockIdx.x * blockDim.x + threadIdx.x;
  if (i < n) wh[i] = (_Float16)w[i];
}

// ---------------------------------------------------------------------------
// Generic neighbor mean: mask = (adj>0) & ~eye, divide by row degree. f16 out.
// ---------------------------------------------------------------------------
__global__ __launch_bounds__(256)
void neigh_kernel(const float* __restrict__ x, const int* __restrict__ adj,
                  _Float16* __restrict__ nbh) {
  const int i = blockIdx.x, b = blockIdx.y, tid = threadIdx.x;
  const float* xb = x + (size_t)b * SS * DD;
  const int* arow = adj + ((size_t)b * SS + i) * SS;
  float acc[16];
#pragma unroll
  for (int c = 0; c < 16; ++c) acc[c] = 0.f;
  float deg = 0.f;
  for (int j = 0; j < SS; ++j) {
    if (j != i && arow[j] > 0) {
      deg += 1.f;
      const float* xr = xb + (size_t)j * DD;
#pragma unroll
      for (int c = 0; c < 16; ++c) acc[c] += xr[c * 256 + tid];
    }
  }
  float inv = 1.f / deg;
  _Float16* o = nbh + ((size_t)b * SS + i) * DD;
#pragma unroll
  for (int c = 0; c < 16; ++c) o[c * 256 + tid] = (_Float16)(acc[c] * inv);
}

// ---------------------------------------------------------------------------
// sim = (x1f @ x2f^T) / max(n1*n2, 1e-6). One wave -> 16x64 tile, K=4096.
// ---------------------------------------------------------------------------
__global__ __launch_bounds__(32)
void sim_kernel(const _Float16* __restrict__ x1h, const _Float16* __restrict__ x2h,
                const float* __restrict__ n1, const float* __restrict__ n2,
                float* __restrict__ sim) {
  const int b  = blockIdx.z;
  const int m0 = blockIdx.x * 16;
  const int n0 = blockIdx.y * 64;
  const _Float16* A  = x1h + (size_t)b * SS * DD + (size_t)m0 * DD;
  const _Float16* Bp = x2h + (size_t)b * SS * DD + (size_t)n0 * DD;
  v8f acc[4] = {};
  for (int k = 0; k < DD; k += 32) {
    v16h a = load_fragA(A + k, DD);
#pragma unroll
    for (int t = 0; t < 4; ++t) {
      v16h bf = load_fragB(Bp + (size_t)t * 16 * DD + k, DD);
      acc[t] = wmma16(a, bf, acc[t]);
    }
  }
  const int lane = threadIdx.x & 31;
  const int n  = lane & 15;
  const int mb = (lane >> 4) * 8;
  float* out = sim + (size_t)b * SS * SS;
#pragma unroll
  for (int t = 0; t < 4; ++t) {
    int j = n0 + t * 16 + n;
    float nj = n2[b * SS + j];
#pragma unroll
    for (int r = 0; r < 8; ++r) {
      int i = m0 + mb + r;
      float den = fmaxf(n1[b * SS + i] * nj, 1e-6f);
      out[(size_t)i * SS + j] = acc[t][r] / den;
    }
  }
}

// ---------------------------------------------------------------------------
// Softmax over rows of sim (att1) and over columns of sim (att2), f16 out.
// ---------------------------------------------------------------------------
__global__ __launch_bounds__(256)
void softmax_row_kernel(const float* __restrict__ sim, _Float16* __restrict__ att) {
  const int i = blockIdx.x, b = blockIdx.y, tid = threadIdx.x;
  const float* row = sim + ((size_t)b * SS + i) * SS;
  float v0 = row[tid], v1 = row[tid + 256];
  __shared__ float sm[256];
  sm[tid] = fmaxf(v0, v1); __syncthreads();
  for (int st = 128; st > 0; st >>= 1) {
    if (tid < st) sm[tid] = fmaxf(sm[tid], sm[tid + st]);
    __syncthreads();
  }
  float m = sm[0]; __syncthreads();
  float e0 = expf(v0 - m), e1 = expf(v1 - m);
  sm[tid] = e0 + e1; __syncthreads();
  for (int st = 128; st > 0; st >>= 1) {
    if (tid < st) sm[tid] += sm[tid + st];
    __syncthreads();
  }
  float inv = 1.f / sm[0];
  _Float16* o = att + ((size_t)b * SS + i) * SS;
  o[tid]       = (_Float16)(e0 * inv);
  o[tid + 256] = (_Float16)(e1 * inv);
}

__global__ __launch_bounds__(256)
void softmax_col_kernel(const float* __restrict__ sim, _Float16* __restrict__ att) {
  const int i = blockIdx.x, b = blockIdx.y, tid = threadIdx.x;
  const float* col = sim + (size_t)b * SS * SS + i;
  float v0 = col[(size_t)tid * SS], v1 = col[(size_t)(tid + 256) * SS];
  __shared__ float sm[256];
  sm[tid] = fmaxf(v0, v1); __syncthreads();
  for (int st = 128; st > 0; st >>= 1) {
    if (tid < st) sm[tid] = fmaxf(sm[tid], sm[tid + st]);
    __syncthreads();
  }
  float m = sm[0]; __syncthreads();
  float e0 = expf(v0 - m), e1 = expf(v1 - m);
  sm[tid] = e0 + e1; __syncthreads();
  for (int st = 128; st > 0; st >>= 1) {
    if (tid < st) sm[tid] += sm[tid + st];
    __syncthreads();
  }
  float inv = 1.f / sm[0];
  _Float16* o = att + ((size_t)b * SS + i) * SS;
  o[tid]       = (_Float16)(e0 * inv);
  o[tid + 256] = (_Float16)(e1 * inv);
}

// ---------------------------------------------------------------------------
// mu = x_self - att @ x_other   (B operand from transposed f16 copy [b][d][s])
// ---------------------------------------------------------------------------
__global__ __launch_bounds__(32)
void mu_kernel(const _Float16* __restrict__ atth, const _Float16* __restrict__ xTo,
               const _Float16* __restrict__ xs, _Float16* __restrict__ muh) {
  const int b  = blockIdx.z;
  const int m0 = blockIdx.x * 16;   // i
  const int n0 = blockIdx.y * 64;   // d
  const _Float16* A  = atth + (size_t)b * SS * SS + (size_t)m0 * SS;
  const _Float16* Bp = xTo  + (size_t)b * DD * SS + (size_t)n0 * SS;
  v8f acc[4] = {};
  for (int k = 0; k < SS; k += 32) {
    v16h a = load_fragA(A + k, SS);
#pragma unroll
    for (int t = 0; t < 4; ++t) {
      v16h bf = load_fragB(Bp + (size_t)t * 16 * SS + k, SS);
      acc[t] = wmma16(a, bf, acc[t]);
    }
  }
  const int lane = threadIdx.x & 31;
  const int n  = lane & 15;
  const int mb = (lane >> 4) * 8;
  const _Float16* xsb = xs + (size_t)b * SS * DD;
  _Float16* ob = muh + (size_t)b * SS * DD;
#pragma unroll
  for (int t = 0; t < 4; ++t) {
    int d = n0 + t * 16 + n;
#pragma unroll
    for (int r = 0; r < 8; ++r) {
      int i = m0 + mb + r;
      float v = (float)xsb[(size_t)i * DD + d] - acc[t][r];
      ob[(size_t)i * DD + d] = (_Float16)v;
    }
  }
}

// ---------------------------------------------------------------------------
// lin: out[row*1536 + col] = X[32768x512] @ W^T[512x512] + bias (f32 out into
// the concatenated h buffer; caller offsets `out` by the column block).
// ---------------------------------------------------------------------------
__global__ __launch_bounds__(32)
void lin_kernel(const _Float16* __restrict__ X, const _Float16* __restrict__ W,
                const float* __restrict__ bias, float* __restrict__ out) {
  const int m0 = blockIdx.x * 16;
  const int n0 = blockIdx.y * 64;
  const _Float16* A = X + (size_t)m0 * FF;
  v8f acc[4] = {};
  for (int k = 0; k < FF; k += 32) {
    v16h a = load_fragA(A + k, FF);
#pragma unroll
    for (int t = 0; t < 4; ++t) {
      v16h bf = load_fragB(W + (size_t)(n0 + t * 16) * FF + k, FF);
      acc[t] = wmma16(a, bf, acc[t]);
    }
  }
  const int lane = threadIdx.x & 31;
  const int n  = lane & 15;
  const int mb = (lane >> 4) * 8;
#pragma unroll
  for (int t = 0; t < 4; ++t) {
    int col = n0 + t * 16 + n;
    float bv = bias[col];
#pragma unroll
    for (int r = 0; r < 8; ++r) {
      int row = m0 + mb + r;
      out[(size_t)row * C3 + col] = acc[t][r] + bv;
    }
  }
}

// ---------------------------------------------------------------------------
// norm_relu (per-row L2 normalize over 1536 channels, then ReLU) in place.
// ---------------------------------------------------------------------------
__global__ __launch_bounds__(256)
void normrelu_kernel(float* __restrict__ h) {
  const size_t row = blockIdx.x;
  const int tid = threadIdx.x;
  float* r = h + row * C3;
  float v[6]; float ss = 0.f;
#pragma unroll
  for (int c = 0; c < 6; ++c) { v[c] = r[c * 256 + tid]; ss += v[c] * v[c]; }
  __shared__ float sm[256];
  sm[tid] = ss; __syncthreads();
  for (int st = 128; st > 0; st >>= 1) {
    if (tid < st) sm[tid] += sm[tid + st];
    __syncthreads();
  }
  float inv = 1.f / fmaxf(sqrtf(sm[0]), 1e-12f);
#pragma unroll
  for (int c = 0; c < 6; ++c) r[c * 256 + tid] = fmaxf(v[c] * inv, 0.f);
}

// ---------------------------------------------------------------------------
// BN training stats (per tensor h1/h2, per channel), two-level reduction.
// ---------------------------------------------------------------------------
__global__ void zero_kernel(float* __restrict__ p, int n) {
  int i = blockIdx.x * blockDim.x + threadIdx.x;
  if (i < n) p[i] = 0.f;
}

__global__ __launch_bounds__(256)
void stats_kernel(const float* __restrict__ h, float* __restrict__ sums) {
  const size_t row0 = (size_t)blockIdx.x * 128;
  const int tensor = (row0 >= (size_t)ROWS) ? 1 : 0;
  const int tid = threadIdx.x;
  float s[6] = {0, 0, 0, 0, 0, 0}, q[6] = {0, 0, 0, 0, 0, 0};
  for (int rr = 0; rr < 128; ++rr) {
    const float* r = h + (row0 + rr) * C3;
#pragma unroll
    for (int c = 0; c < 6; ++c) {
      float y = r[c * 256 + tid];
      s[c] += y; q[c] += y * y;
    }
  }
  float* base = sums + (size_t)tensor * 2 * C3;
#pragma unroll
  for (int c = 0; c < 6; ++c) {
    atomicAdd(&base[c * 256 + tid], s[c]);
    atomicAdd(&base[C3 + c * 256 + tid], q[c]);
  }
}

__global__ void finalize_kernel(const float* __restrict__ sums,
                                const float* __restrict__ gamma,
                                const float* __restrict__ beta,
                                float* __restrict__ ss) {
  int idx = blockIdx.x * blockDim.x + threadIdx.x;
  if (idx >= 2 * C3) return;
  int tensor = idx / C3, c = idx % C3;
  const float* base = sums + (size_t)tensor * 2 * C3;
  float mean = base[c] / (float)ROWS;
  float var  = base[C3 + c] / (float)ROWS - mean * mean;  // biased
  float inv  = rsqrtf(var + 1e-5f);
  float sc   = gamma[c] * inv;
  ss[(size_t)tensor * 2 * C3 + c]      = sc;
  ss[(size_t)tensor * 2 * C3 + C3 + c] = beta[c] - mean * sc;
}

__global__ __launch_bounds__(256)
void apply_kernel(float* __restrict__ h, const float* __restrict__ ss) {
  size_t base = (size_t)blockIdx.x * 1024 + threadIdx.x;
#pragma unroll
  for (int u = 0; u < 4; ++u) {
    size_t e = base + (size_t)u * 256;
    size_t row = e / C3;
    int c = (int)(e - row * C3);
    int tensor = (row >= (size_t)ROWS) ? 1 : 0;
    const float* b = ss + (size_t)tensor * 2 * C3;
    h[e] = h[e] * b[c] + b[C3 + c];
  }
}

// ---------------------------------------------------------------------------
extern "C" void kernel_launch(void* const* d_in, const int* in_sizes, int n_in,
                              void* d_out, int out_size, void* d_ws, size_t ws_size,
                              hipStream_t stream) {
  const float* x1    = (const float*)d_in[0];
  const float* x2    = (const float*)d_in[1];
  const int*   adj1  = (const int*)d_in[2];
  const int*   adj2  = (const int*)d_in[3];
  const float* Wx    = (const float*)d_in[4];
  const float* bx    = (const float*)d_in[5];
  const float* Wn    = (const float*)d_in[6];
  const float* bnb   = (const float*)d_in[7];
  const float* Wr    = (const float*)d_in[8];
  const float* br    = (const float*)d_in[9];
  const float* gamma = (const float*)d_in[10];
  const float* beta  = (const float*)d_in[11];
  float* out = (float*)d_out;

  char* w = (char*)d_ws;
  size_t off = 0;
  auto carve = [&](size_t bytes) -> void* {
    void* p = w + off;
    off = (off + bytes + 255) & ~(size_t)255;
    return p;
  };
  _Float16* x1h  = (_Float16*)carve((size_t)NBS * DD * sizeof(_Float16));
  _Float16* x2h  = (_Float16*)carve((size_t)NBS * DD * sizeof(_Float16));
  _Float16* x1T  = (_Float16*)carve((size_t)NBS * DD * sizeof(_Float16));
  _Float16* x2T  = (_Float16*)carve((size_t)NBS * DD * sizeof(_Float16));
  _Float16* nb1h = (_Float16*)carve((size_t)NBS * DD * sizeof(_Float16));
  _Float16* nb2h = (_Float16*)carve((size_t)NBS * DD * sizeof(_Float16));
  _Float16* mu1h = (_Float16*)carve((size_t)NBS * DD * sizeof(_Float16));
  _Float16* mu2h = (_Float16*)carve((size_t)NBS * DD * sizeof(_Float16));
  float*    sim  = (float*)   carve((size_t)BB * SS * SS * sizeof(float));
  _Float16* att1 = (_Float16*)carve((size_t)BB * SS * SS * sizeof(_Float16));
  _Float16* att2 = (_Float16*)carve((size_t)BB * SS * SS * sizeof(_Float16));
  _Float16* Wxh  = (_Float16*)carve((size_t)OUTC * FF * sizeof(_Float16));
  _Float16* Wnh  = (_Float16*)carve((size_t)OUTC * FF * sizeof(_Float16));
  _Float16* Wrh  = (_Float16*)carve((size_t)OUTC * FF * sizeof(_Float16));
  float*    n1   = (float*)   carve((size_t)NBS * sizeof(float));
  float*    n2   = (float*)   carve((size_t)NBS * sizeof(float));
  float*    stats = (float*)  carve((size_t)4 * C3 * sizeof(float));
  float*    ssbuf = (float*)  carve((size_t)4 * C3 * sizeof(float));
  (void)in_sizes; (void)n_in; (void)out_size; (void)ws_size;

  dim3 gbs(SS, BB);

  convx_kernel<<<gbs, 256, 0, stream>>>(x1, x1h, x1T, n1);
  convx_kernel<<<gbs, 256, 0, stream>>>(x2, x2h, x2T, n2);
  convw_kernel<<<(OUTC * FF + 255) / 256, 256, 0, stream>>>(Wx, Wxh, OUTC * FF);
  convw_kernel<<<(OUTC * FF + 255) / 256, 256, 0, stream>>>(Wn, Wnh, OUTC * FF);
  convw_kernel<<<(OUTC * FF + 255) / 256, 256, 0, stream>>>(Wr, Wrh, OUTC * FF);

  neigh_kernel<<<gbs, 256, 0, stream>>>(x1, adj1, nb1h);
  neigh_kernel<<<gbs, 256, 0, stream>>>(x2, adj2, nb2h);

  sim_kernel<<<dim3(SS / 16, SS / 64, BB), 32, 0, stream>>>(x1h, x2h, n1, n2, sim);

  softmax_row_kernel<<<gbs, 256, 0, stream>>>(sim, att1);
  softmax_col_kernel<<<gbs, 256, 0, stream>>>(sim, att2);

  mu_kernel<<<dim3(SS / 16, DD / 64, BB), 32, 0, stream>>>(att1, x2T, x1h, mu1h);
  mu_kernel<<<dim3(SS / 16, DD / 64, BB), 32, 0, stream>>>(att2, x1T, x2h, mu2h);

  float* h1 = out;
  float* h2 = out + (size_t)ROWS * C3;
  dim3 gl(ROWS / 16, OUTC / 64);
  lin_kernel<<<gl, 32, 0, stream>>>(x1h,  Wxh, bx,  h1);
  lin_kernel<<<gl, 32, 0, stream>>>(nb1h, Wnh, bnb, h1 + OUTC);
  lin_kernel<<<gl, 32, 0, stream>>>(mu1h, Wrh, br,  h1 + 2 * OUTC);
  lin_kernel<<<gl, 32, 0, stream>>>(x2h,  Wxh, bx,  h2);
  lin_kernel<<<gl, 32, 0, stream>>>(nb2h, Wnh, bnb, h2 + OUTC);
  lin_kernel<<<gl, 32, 0, stream>>>(mu2h, Wrh, br,  h2 + 2 * OUTC);

  normrelu_kernel<<<2 * ROWS, 256, 0, stream>>>(out);

  zero_kernel<<<(4 * C3 + 255) / 256, 256, 0, stream>>>(stats, 4 * C3);
  stats_kernel<<<2 * ROWS / 128, 256, 0, stream>>>(out, stats);
  finalize_kernel<<<(2 * C3 + 255) / 256, 256, 0, stream>>>(stats, gamma, beta, ssbuf);
  apply_kernel<<<(unsigned)(((size_t)2 * ROWS * C3) / 1024), 256, 0, stream>>>(out, ssbuf);
}
